// SelfAttention_49108656062900
// MI455X (gfx1250) — matrix operand order repaired and checked by reading, not compiled, not run on hardware
//
#include <hip/hip_runtime.h>
#include <hip/hip_bf16.h>

// ---------------------------------------------------------------------------
// Self-attention forward for B=16, L=1024, D=1024, H=16, HD=64 on gfx1250.
// bf16 WMMA (v_wmma_f32_16x16x32_bf16), fp32 accumulate.
// Prepass: weights -> bf16, mask -> bitmask (32x less mask traffic).
// ---------------------------------------------------------------------------

#define BB 16
#define LL 1024
#define DD 1024
#define HH 16
#define HD 64
#define MM (BB * LL)          // 16384 rows in the projection GEMMs
#define LW (LL / 32)          // mask words per row

typedef __attribute__((ext_vector_type(16))) __bf16 v16bf;
typedef __attribute__((ext_vector_type(8)))  __bf16 v8bf;
typedef __attribute__((ext_vector_type(8)))  float  v8f;

// Pack two floats into two bf16 (round-half-up) with a single v_perm_b32.
static __device__ __forceinline__ unsigned pack2bf(float a, float b) {
    unsigned ua = __builtin_bit_cast(unsigned, a) + 0x8000u;
    unsigned ub = __builtin_bit_cast(unsigned, b) + 0x8000u;
    // result bytes: [ub.b3, ub.b2, ua.b3, ua.b2] -> (bf(b)<<16)|bf(a)
    return __builtin_amdgcn_perm(ub, ua, 0x07060302u);
}

static __device__ __forceinline__ __bf16 f2bf(float f) {
    unsigned u = (__builtin_bit_cast(unsigned, f) + 0x8000u) >> 16;
    unsigned short s = (unsigned short)u;
    return __builtin_bit_cast(__bf16, s);
}

union BF16x16w { unsigned w[8]; v16bf v; };
union BF16x16p { struct { v8bf lo, hi; } s; v16bf v; };

// 8 fp32 -> 4 packed bf16 words
static __device__ __forceinline__ void cvt8(const float* __restrict__ p, unsigned* w) {
    const float4* q = (const float4*)p;
    float4 x = q[0], y = q[1];
    w[0] = pack2bf(x.x, x.y);
    w[1] = pack2bf(x.z, x.w);
    w[2] = pack2bf(y.x, y.y);
    w[3] = pack2bf(y.z, y.w);
}

// A-fragment (16x32, M x K): lane holds row = lane%16, 16 contiguous K values
// starting at 16h.  fp32 source (converted on the fly).
static __device__ __forceinline__ v16bf load_a_f32(const float* __restrict__ row, int half) {
    BF16x16w u;
    cvt8(row + 16 * half, u.w);
    cvt8(row + 16 * half + 8, u.w + 4);
    return u.v;
}

static __device__ __forceinline__ v16bf load_a_bf16(const __bf16* __restrict__ row, int half) {
    BF16x16p u;
    u.s.lo = *(const v8bf*)(row + 16 * half);
    u.s.hi = *(const v8bf*)(row + 16 * half + 8);
    return u.v;
}

// B-fragment (32x16, K x N): lane holds col = lane%16, 16 contiguous K values
// starting at 16*(lane/16).  Pointer already includes the 16*half offset.
static __device__ __forceinline__ v16bf load_b_bf16(const __bf16* __restrict__ p) {
    BF16x16p u;
    u.s.lo = *(const v8bf*)(p);
    u.s.hi = *(const v8bf*)(p + 8);
    return u.v;
}

static __device__ __forceinline__ v8f wmma_bf16(v16bf a, v16bf b, v8f c) {
    return __builtin_amdgcn_wmma_f32_16x16x32_bf16(
        /*neg_a=*/false, a, /*neg_b=*/false, b,
        /*c_mod=*/(short)0, c, /*reuse_a=*/false, /*reuse_b=*/false);
}

// ---------------------------------------------------------------------------
// Prepass A: weight matrix fp32 -> bf16 (1M elements, 2 per thread).
// ---------------------------------------------------------------------------
__global__ __launch_bounds__(256) void cvt_w_kernel(const float* __restrict__ W,
                                                    __bf16* __restrict__ out) {
    const int i = blockIdx.x * blockDim.x + threadIdx.x;   // over (D*D)/2
    const float2 f = ((const float2*)W)[i];
    ((unsigned*)out)[i] = pack2bf(f.x, f.y);
}

// ---------------------------------------------------------------------------
// Prepass B: int32 mask [B,L,L] -> bitmask u32 [B,L,L/32] via wave32 ballot.
// ---------------------------------------------------------------------------
__global__ __launch_bounds__(256) void mask_bits_kernel(const int* __restrict__ mask,
                                                        unsigned* __restrict__ bits) {
    const size_t i = (size_t)blockIdx.x * blockDim.x + threadIdx.x; // over B*L*L
    const int pred = (mask[i] != 0);
    const unsigned w = (unsigned)__ballot(pred);  // wave32: bit = lane = column%32
    if ((threadIdx.x & 31) == 0) bits[i >> 5] = w;
}

// ---------------------------------------------------------------------------
// Kernel 1: projection GEMM  C = A(f32)[M,K] @ Wb(bf16)[N,K]^T + bias.
// MODE 0: out[(b*H+h)*L + l][hd]   (head layout, for Q and K)
// MODE 1: out[(b*H+h)*HD + hd][l]  (transposed head layout, for V)
// One wave computes a 16x64 strip of C (4 accumulators share the A fragment).
// ---------------------------------------------------------------------------
template <int MODE>
__global__ __launch_bounds__(256) void proj_kernel(const float* __restrict__ A,
                                                   const __bf16* __restrict__ Wb,
                                                   const float* __restrict__ bias,
                                                   __bf16* __restrict__ out) {
    const int wave = (blockIdx.x * blockDim.x + threadIdx.x) >> 5;
    const int lane = threadIdx.x & 31;
    const int half = lane >> 4;
    const int ln   = lane & 15;

    const int mt = wave >> 4;        // 0..1023  (M tile of 16)
    const int n0 = (wave & 15) << 6; // 0,64,...,960

    const float* arow = A + (size_t)(mt * 16 + ln) * DD;

    v8f acc[4] = {};
    for (int k0 = 0; k0 < DD; k0 += 32) {
        v16bf av = load_a_f32(arow + k0, half);
#pragma unroll
        for (int t = 0; t < 4; ++t) {
            const __bf16* wrow = Wb + (size_t)(n0 + t * 16 + ln) * DD + k0 + 16 * half;
            acc[t] = wmma_bf16(av, load_b_bf16(wrow), acc[t]);
        }
    }

#pragma unroll
    for (int t = 0; t < 4; ++t) {
        const int n  = n0 + t * 16 + ln;
        const int h  = n >> 6;
        const int hd = n & 63;
        const float bv = bias[n];
#pragma unroll
        for (int j = 0; j < 8; ++j) {
            const int m    = mt * 16 + j + 8 * half;
            const int b    = m >> 10;
            const int lseq = m & 1023;
            const __bf16 val = f2bf(acc[t][j] + bv);
            if (MODE == 0)
                out[((size_t)(b * HH + h) * LL + lseq) * HD + hd] = val;
            else
                out[((size_t)(b * HH + h) * HD + hd) * LL + lseq] = val;
        }
    }
}

// ---------------------------------------------------------------------------
// Kernel 2: flash attention.  One wave owns a 16-row Q tile of one (b,h) and
// streams over KV in 32-wide chunks with online softmax + bitmask.
// ---------------------------------------------------------------------------
__global__ __launch_bounds__(256) void flash_attn_kernel(const __bf16* __restrict__ Qh,
                                                         const __bf16* __restrict__ Kh,
                                                         const __bf16* __restrict__ VhT,
                                                         const unsigned* __restrict__ mbits,
                                                         __bf16* __restrict__ Y) {
    __shared__ __align__(32) __bf16 ldsP[8][16 * 32]; // per-wave P transpose buffer

    const int wslot = threadIdx.x >> 5;
    const int gw    = blockIdx.x * 8 + wslot;
    const int lane  = threadIdx.x & 31;
    const int half  = lane >> 4;
    const int ln    = lane & 15;

    const int qt = gw & 63;   // q tile (16 rows each)
    const int bh = gw >> 6;   // 0..255
    const int b  = bh >> 4;
    const int h  = bh & 15;

    // Preload Q A-fragments for d-blocks [0,32) and [32,64)
    const __bf16* qrow = Qh + ((size_t)bh * LL + qt * 16 + ln) * HD;
    const v16bf qa0 = load_a_bf16(qrow, half);
    const v16bf qa1 = load_a_bf16(qrow + 32, half);

    const __bf16* kbase = Kh + (size_t)bh * LL * HD;
    const __bf16* vbase = VhT + (size_t)bh * HD * LL;
    const unsigned* mrow = mbits + ((size_t)b * LL + qt * 16 + 8 * half) * LW;
    __bf16* pw = &ldsP[wslot][0];

    v8f o[4] = {};
    float mrv[8], lrv[8];
#pragma unroll
    for (int j = 0; j < 8; ++j) { mrv[j] = -1e30f; lrv[j] = 0.0f; }

    const float scale = 0.125f; // 1/sqrt(64)

    for (int kv0 = 0; kv0 < LL; kv0 += 32) {
        // ---- S = Q K^T for two 16-col tiles ----
        const __bf16* k0p = kbase + (size_t)(kv0 + ln) * HD + 16 * half;
        const __bf16* k1p = k0p + 16 * HD;
        v8f s0 = {}, s1 = {};
        s0 = wmma_bf16(qa0, load_b_bf16(k0p), s0);
        s0 = wmma_bf16(qa1, load_b_bf16(k0p + 32), s0);
        s1 = wmma_bf16(qa0, load_b_bf16(k1p), s1);
        s1 = wmma_bf16(qa1, load_b_bf16(k1p + 32), s1);

        if (kv0 + 32 < LL) {  // prefetch next K / V chunk (global_prefetch_b8)
            __builtin_prefetch(kbase + (size_t)(kv0 + 32 + ln) * HD, 0, 0);
            __builtin_prefetch(vbase + (size_t)ln * LL + kv0 + 32, 0, 0);
        }

        // ---- bitmask words for this chunk (one u32 covers all 32 columns) ----
        unsigned mw[8];
#pragma unroll
        for (int j = 0; j < 8; ++j) mw[j] = mrow[(size_t)j * LW + (kv0 >> 5)];

        // ---- mask + online softmax ----
        float p0[8], p1[8], alpha[8];
#pragma unroll
        for (int j = 0; j < 8; ++j) {
            float x0 = ((mw[j] >> ln) & 1u)        ? s0[j] * scale : -1e30f;
            float x1 = ((mw[j] >> (16 + ln)) & 1u) ? s1[j] * scale : -1e30f;

            float rm = fmaxf(x0, x1);
            rm = fmaxf(rm, __shfl_xor(rm, 1, 32));
            rm = fmaxf(rm, __shfl_xor(rm, 2, 32));
            rm = fmaxf(rm, __shfl_xor(rm, 4, 32));
            rm = fmaxf(rm, __shfl_xor(rm, 8, 32));

            const float mnew = fmaxf(mrv[j], rm);
            alpha[j] = __expf(mrv[j] - mnew);
            mrv[j]   = mnew;
            p0[j] = __expf(x0 - mnew);
            p1[j] = __expf(x1 - mnew);

            float rs = p0[j] + p1[j];
            rs += __shfl_xor(rs, 1, 32);
            rs += __shfl_xor(rs, 2, 32);
            rs += __shfl_xor(rs, 4, 32);
            rs += __shfl_xor(rs, 8, 32);
            lrv[j] = lrv[j] * alpha[j] + rs;
        }
#pragma unroll
        for (int t = 0; t < 4; ++t)
#pragma unroll
            for (int j = 0; j < 8; ++j) o[t][j] *= alpha[j];

        // ---- transpose P (C layout -> A layout) through private LDS slice ----
#pragma unroll
        for (int j = 0; j < 8; ++j) {
            pw[(j + 8 * half) * 32 + ln]      = f2bf(p0[j]);
            pw[(j + 8 * half) * 32 + 16 + ln] = f2bf(p1[j]);
        }
        asm volatile("" ::: "memory"); // keep ds stores before ds loads (same-wave LDS is in-order)
        const v16bf pa = load_a_bf16(pw + ln * 32, half);

        // ---- O += P @ V  (V stored transposed: contiguous over kv) ----
#pragma unroll
        for (int t = 0; t < 4; ++t) {
            const __bf16* vp = vbase + (size_t)(t * 16 + ln) * LL + kv0 + 16 * half;
            o[t] = wmma_bf16(pa, load_b_bf16(vp), o[t]);
        }
    }

    // ---- normalize and store to Y[B,L,D] (bf16) ----
#pragma unroll
    for (int j = 0; j < 8; ++j) {
        const float inv = 1.0f / lrv[j];
        const int q = qt * 16 + j + 8 * half;
#pragma unroll
        for (int t = 0; t < 4; ++t) {
            Y[((size_t)(b * LL + q)) * DD + h * HD + t * 16 + ln] = f2bf(o[t][j] * inv);
        }
    }
}

// ---------------------------------------------------------------------------
// Kernel 3: output projection  out = Y(bf16)[M,K] @ Wp(bf16)[N,K]^T + bp (f32).
// ---------------------------------------------------------------------------
__global__ __launch_bounds__(256) void out_proj_kernel(const __bf16* __restrict__ Y,
                                                       const __bf16* __restrict__ Wb,
                                                       const float* __restrict__ bias,
                                                       float* __restrict__ out) {
    const int wave = (blockIdx.x * blockDim.x + threadIdx.x) >> 5;
    const int lane = threadIdx.x & 31;
    const int half = lane >> 4;
    const int ln   = lane & 15;

    const int mt = wave >> 4;
    const int n0 = (wave & 15) << 6;

    const __bf16* arow = Y + (size_t)(mt * 16 + ln) * DD;

    v8f acc[4] = {};
    for (int k0 = 0; k0 < DD; k0 += 32) {
        v16bf av = load_a_bf16(arow + k0, half);
#pragma unroll
        for (int t = 0; t < 4; ++t) {
            const __bf16* wrow = Wb + (size_t)(n0 + t * 16 + ln) * DD + k0 + 16 * half;
            acc[t] = wmma_bf16(av, load_b_bf16(wrow), acc[t]);
        }
    }

#pragma unroll
    for (int t = 0; t < 4; ++t) {
        const int n = n0 + t * 16 + ln;
        const float bv = bias[n];
#pragma unroll
        for (int j = 0; j < 8; ++j) {
            const int m = mt * 16 + j + 8 * half;
            out[(size_t)m * DD + n] = acc[t][j] + bv;
        }
    }
}

// ---------------------------------------------------------------------------
// Launch
// ---------------------------------------------------------------------------
extern "C" void kernel_launch(void* const* d_in, const int* in_sizes, int n_in,
                              void* d_out, int out_size, void* d_ws, size_t ws_size,
                              hipStream_t stream) {
    const float* key   = (const float*)d_in[0];
    const float* value = (const float*)d_in[1];
    const float* query = (const float*)d_in[2];
    const int*   masks = (const int*)d_in[3];
    const float* Wk = (const float*)d_in[4];
    const float* bk = (const float*)d_in[5];
    const float* Wq = (const float*)d_in[6];
    const float* bq = (const float*)d_in[7];
    const float* Wv = (const float*)d_in[8];
    const float* bv = (const float*)d_in[9];
    const float* Wp = (const float*)d_in[10];
    const float* bp = (const float*)d_in[11];
    float* out = (float*)d_out;

    // Workspace layout (bytes):
    //   Qh/Kh/VhT/Yb : 4 x 33,554,432 bf16 buffers
    //   Wq/Wk/Wv/Wp  : 4 x  2,097,152 bf16 weights
    //   mask bits    :      2,097,152
    const size_t elems = (size_t)BB * HH * LL * HD;   // 16,777,216
    char* base = (char*)d_ws;
    __bf16* Qh  = (__bf16*)(base);
    __bf16* Kh  = (__bf16*)(base + 2 * elems);
    __bf16* VhT = (__bf16*)(base + 4 * elems);
    __bf16* Yb  = (__bf16*)(base + 6 * elems);
    __bf16* Wqb = (__bf16*)(base + 8 * elems);
    __bf16* Wkb = (__bf16*)(base + 8 * elems + 2u * DD * DD);
    __bf16* Wvb = (__bf16*)(base + 8 * elems + 4u * DD * DD);
    __bf16* Wpb = (__bf16*)(base + 8 * elems + 6u * DD * DD);
    unsigned* mbits = (unsigned*)(base + 8 * elems + 8u * DD * DD);

    const dim3 block(256);
    const dim3 wcvt_grid((DD * DD / 2) / 256);                 // 2048 blocks
    const dim3 mask_grid(((size_t)BB * LL * LL) / 256);        // 65536 blocks
    const dim3 gemm_grid((MM / 16) * (DD / 64) / 8);           // 2048 blocks
    const dim3 attn_grid((BB * HH * (LL / 16)) / 8);           // 2048 blocks

    cvt_w_kernel<<<wcvt_grid, block, 0, stream>>>(Wq, Wqb);
    cvt_w_kernel<<<wcvt_grid, block, 0, stream>>>(Wk, Wkb);
    cvt_w_kernel<<<wcvt_grid, block, 0, stream>>>(Wv, Wvb);
    cvt_w_kernel<<<wcvt_grid, block, 0, stream>>>(Wp, Wpb);
    mask_bits_kernel<<<mask_grid, block, 0, stream>>>(masks, mbits);

    proj_kernel<0><<<gemm_grid, block, 0, stream>>>(query, Wqb, bq, Qh);
    proj_kernel<0><<<gemm_grid, block, 0, stream>>>(key,   Wkb, bk, Kh);
    proj_kernel<1><<<gemm_grid, block, 0, stream>>>(value, Wvb, bv, VhT);
    flash_attn_kernel<<<attn_grid, block, 0, stream>>>(Qh, Kh, VhT, mbits, Yb);
    out_proj_kernel<<<gemm_grid, block, 0, stream>>>(Yb, Wpb, bp, out);
}
